// Block_11458972745854
// MI455X (gfx1250) — matrix operand order repaired and checked by reading, not compiled, not run on hardware
//
#include <hip/hip_runtime.h>
#include <hip/hip_bf16.h>

// ---------------------------------------------------------------------------
// Types / helpers
// ---------------------------------------------------------------------------
typedef __attribute__((ext_vector_type(16))) __bf16 bf16x16;
typedef __attribute__((ext_vector_type(8)))  float  f32x8;

union Frag {
    bf16x16 v;
    uint4   q[2];
    unsigned int   u[8];
    unsigned short s[16];
};

__device__ __forceinline__ unsigned short f2bf(float f) {
    unsigned int u = __float_as_uint(f);
    unsigned int r = u + 0x7FFFu + ((u >> 16) & 1u);   // round to nearest even
    return (unsigned short)(r >> 16);
}

// ---- CDNA5 async Global->LDS DMA (ASYNCcnt path), via inline asm ----------
// VDST VGPR holds the wave-relative LDS byte address = low 32 bits of the
// generic pointer (aperture lives in the high half).
__device__ __forceinline__ void async_copy_b128(void* lds, const void* g) {
    unsigned            l  = (unsigned)(unsigned long long)lds;
    unsigned long long  ga = (unsigned long long)g;
    asm volatile("global_load_async_to_lds_b128 %0, %1, off"
                 :: "v"(l), "v"(ga) : "memory");
}
__device__ __forceinline__ void wait_async0() {
    asm volatile("s_wait_asynccnt 0x0" ::: "memory");
}

#define ROWS   4096     // B*T
#define CMODEL 1024
#define C3     3072
#define DFF    4096
#define TSEQ   2048

// ---------------------------------------------------------------------------
// LayerNorm: fp32 row -> bf16 row.  One block (256 thr) per row of 1024.
// ---------------------------------------------------------------------------
__launch_bounds__(256)
__global__ void ln_bf16_kernel(const float* __restrict__ x,
                               const float* __restrict__ g,
                               const float* __restrict__ b,
                               unsigned short* __restrict__ out) {
    const int row = blockIdx.x;
    const float* xr = x + (size_t)row * CMODEL;
    const int t = threadIdx.x, lane = t & 31, wave = t >> 5;

    float vals[4], s = 0.f, s2 = 0.f;
#pragma unroll
    for (int i = 0; i < 4; ++i) {
        float v = xr[t + i * 256];
        vals[i] = v; s += v; s2 += v * v;
    }
#pragma unroll
    for (int off = 1; off < 32; off <<= 1) {
        s  += __shfl_xor(s,  off, 32);
        s2 += __shfl_xor(s2, off, 32);
    }
    __shared__ float ss[8], ss2[8];
    if (lane == 0) { ss[wave] = s; ss2[wave] = s2; }
    __syncthreads();
    float ts = 0.f, ts2 = 0.f;
#pragma unroll
    for (int w = 0; w < 8; ++w) { ts += ss[w]; ts2 += ss2[w]; }
    const float mu   = ts * (1.0f / CMODEL);
    const float var  = ts2 * (1.0f / CMODEL) - mu * mu;
    const float rstd = rsqrtf(var + 1e-5f);
#pragma unroll
    for (int i = 0; i < 4; ++i) {
        int c = t + i * 256;
        out[(size_t)row * CMODEL + c] = f2bf((vals[i] - mu) * rstd * g[c] + b[c]);
    }
}

// ---------------------------------------------------------------------------
// Transpose + convert: W[K][N] fp32 -> Wt[N][K] bf16.  32x32 LDS tile.
// ---------------------------------------------------------------------------
__launch_bounds__(256)
__global__ void transpose_bf16_kernel(const float* __restrict__ w,
                                      unsigned short* __restrict__ wt,
                                      int K, int N) {
    __shared__ unsigned short tile[32][33];
    const int n0 = blockIdx.x * 32, k0 = blockIdx.y * 32;
    const int tx = threadIdx.x & 31, ty = threadIdx.x >> 5;   // 32 x 8
#pragma unroll
    for (int i = 0; i < 4; ++i) {
        int k = ty + i * 8;
        tile[k][tx] = f2bf(w[(size_t)(k0 + k) * N + (n0 + tx)]);
    }
    __syncthreads();
#pragma unroll
    for (int i = 0; i < 4; ++i) {
        int r = ty + i * 8;   // local n
        wt[(size_t)(n0 + r) * K + (k0 + tx)] = tile[tx][r];
    }
}

// ---------------------------------------------------------------------------
// GEMM: C[M][N] = A[M][K] (bf16, row-major) * Bt[N][K]^T (bf16, row-major)
// Block tile 128x128, BK=32, 256 threads = 8 waves, each wave 64x32.
// LDS tiles double-buffered and filled by async Global->LDS DMA.
// Epilogues: 0 = bf16 store, 1 = +residual -> fp32 store, 2 = GELU -> bf16.
// ---------------------------------------------------------------------------
#define BM 128
#define BN 128
#define BK 32
#define LDT 40          // padded LDS row stride in ushorts (80B)

__launch_bounds__(256)
__global__ void gemm_bf16_wmma_kernel(const unsigned short* __restrict__ A,
                                      const unsigned short* __restrict__ Bt,
                                      int M, int N, int K, int epilogue,
                                      unsigned short* __restrict__ outb,
                                      float* __restrict__ outf,
                                      const float* __restrict__ residual) {
    __shared__ __align__(16) unsigned short As[2][BM * LDT];
    __shared__ __align__(16) unsigned short Bs[2][BN * LDT];

    const int t = threadIdx.x;
    const int wave = t >> 5, lane = t & 31, hi = lane >> 4, lm = lane & 15;
    const int bm = blockIdx.y * BM, bn = blockIdx.x * BN;
    const int wm = (wave >> 2) * 64, wn = (wave & 3) * 32;
    const int lrow = t >> 1, lhalf = t & 1;      // tile-fill coords

    f32x8 acc[4][2];
#pragma unroll
    for (int i = 0; i < 4; ++i)
#pragma unroll
        for (int j = 0; j < 2; ++j)
#pragma unroll
            for (int r = 0; r < 8; ++r) acc[i][j][r] = 0.0f;

    const unsigned short* gA = A  + (size_t)(bm + lrow) * K + lhalf * 16;
    const unsigned short* gB = Bt + (size_t)(bn + lrow) * K + lhalf * 16;
    const int ldso = lrow * LDT + lhalf * 16;

    // prologue: async-fill buffer 0 with tile k0=0
    async_copy_b128(&As[0][ldso],     gA);
    async_copy_b128(&As[0][ldso + 8], gA + 8);
    async_copy_b128(&Bs[0][ldso],     gB);
    async_copy_b128(&Bs[0][ldso + 8], gB + 8);
    wait_async0();
    __syncthreads();

    int cur = 0;
    for (int k0 = 0; k0 < K; k0 += BK) {
        // ---- kick off DMA for the next tile into the other buffer ----
        if (k0 + BK < K) {
            int nb = cur ^ 1;
            async_copy_b128(&As[nb][ldso],     gA + k0 + BK);
            async_copy_b128(&As[nb][ldso + 8], gA + k0 + BK + 8);
            async_copy_b128(&Bs[nb][ldso],     gB + k0 + BK);
            async_copy_b128(&Bs[nb][ldso + 8], gB + k0 + BK + 8);
            if (k0 + 2 * BK < K) {               // warm L2 for tile after next
                __builtin_prefetch(gA + k0 + 2 * BK, 0, 1);
                __builtin_prefetch(gB + k0 + 2 * BK, 0, 1);
            }
        }

        // ---- fragments (exact wave32 WMMA VGPR layouts) ----
        Frag af[4], bfr[2];
#pragma unroll
        for (int mt = 0; mt < 4; ++mt) {
            int base = (wm + mt * 16 + lm) * LDT;
            af[mt].q[0] = *(const uint4*)&As[cur][base + hi * 8];        // K 0..7  (+8*hi)
            af[mt].q[1] = *(const uint4*)&As[cur][base + 16 + hi * 8];   // K 16..23(+8*hi)
        }
#pragma unroll
        for (int nt = 0; nt < 2; ++nt) {
            int base = (wn + nt * 16 + lm) * LDT;
            bfr[nt].q[0] = *(const uint4*)&Bs[cur][base + hi * 8];       // K 0..7  (+8*hi)
            bfr[nt].q[1] = *(const uint4*)&Bs[cur][base + 16 + hi * 8];  // K 16..23(+8*hi)
        }
#pragma unroll
        for (int mt = 0; mt < 4; ++mt)
#pragma unroll
            for (int nt = 0; nt < 2; ++nt)
                acc[mt][nt] = __builtin_amdgcn_wmma_f32_16x16x32_bf16(
                    false, af[mt].v, false, bfr[nt].v,
                    (short)0, acc[mt][nt], false, false);

        wait_async0();        // next tile's DMA done (overlapped with WMMAs)
        __syncthreads();      // all waves done reading cur / writing next
        cur ^= 1;
    }

    // ---- epilogue: C layout -> lane lm holds col N, VGPR r holds row r+8*hi ----
#pragma unroll
    for (int mt = 0; mt < 4; ++mt)
#pragma unroll
        for (int nt = 0; nt < 2; ++nt)
#pragma unroll
            for (int r = 0; r < 8; ++r) {
                int row = bm + wm + mt * 16 + r + hi * 8;
                int col = bn + wn + nt * 16 + lm;
                float v = acc[mt][nt][r];
                size_t idx = (size_t)row * N + col;
                if (epilogue == 0) {
                    outb[idx] = f2bf(v);
                } else if (epilogue == 1) {
                    outf[idx] = v + residual[idx];
                } else {  // exact GELU
                    float gl = 0.5f * v * (1.0f + erff(v * 0.70710678118654752f));
                    outb[idx] = f2bf(gl);
                }
            }
}

// ---------------------------------------------------------------------------
// Flash attention (causal).  Block = 128 threads = 4 waves, 64-query tile.
// qkv bf16 [B*T][3072]  (cols: [0,1024)=Q, [1024,2048)=K, [2048,3072)=V,
// within each: col = h*64 + d).   out bf16 [B*T][1024].
// ---------------------------------------------------------------------------
#define KLD 72          // K tile row stride (keys-major, 64 d + pad)
#define VLD 40          // V^T tile row stride (d-major, 32 keys + pad)
#define PLD 40

__launch_bounds__(128)
__global__ void attn_kernel(const unsigned short* __restrict__ qkv,
                            unsigned short* __restrict__ out) {
    const int qb = blockIdx.x & 31;
    const int h  = (blockIdx.x >> 5) & 15;
    const int b  = blockIdx.x >> 9;
    const int q0 = qb * 64;

    const int t = threadIdx.x, wave = t >> 5, lane = t & 31;
    const int hi = lane >> 4, lm = lane & 15;

    __shared__ __align__(16) unsigned short Ks[32 * KLD];
    __shared__ __align__(16) unsigned short Vs[64 * VLD];
    __shared__ __align__(16) unsigned short Ps[4][16 * PLD];

    const unsigned short* Qbase = qkv + (size_t)(b * TSEQ) * C3 + h * 64;
    const unsigned short* Kbase = Qbase + 1024;
    const unsigned short* Vbase = Qbase + 2048;

    // Q fragments (A layout, rows = this wave's 16 queries), d split 32+32
    Frag qf[2];
    {
        const unsigned short* qp = Qbase + (size_t)(q0 + wave * 16 + lm) * C3;
#pragma unroll
        for (int c = 0; c < 2; ++c) {
            qf[c].q[0] = *(const uint4*)(qp + c * 32 + hi * 8);
            qf[c].q[1] = *(const uint4*)(qp + c * 32 + 16 + hi * 8);
        }
    }

    f32x8 o[4];
#pragma unroll
    for (int d = 0; d < 4; ++d)
#pragma unroll
        for (int r = 0; r < 8; ++r) o[d][r] = 0.0f;
    float mrow[8], lrow[8];
#pragma unroll
    for (int r = 0; r < 8; ++r) { mrow[r] = -3.0e38f; lrow[r] = 0.0f; }

    const int qlast = q0 + wave * 16 + 15;

    for (int kv0 = 0; kv0 < q0 + 64; kv0 += 32) {
        // ---- K tile [32 keys][64 d]: straight copy via async DMA ----
        for (int i = t; i < 256; i += 128) {
            int row = i >> 3, j = i & 7;
            async_copy_b128(&Ks[row * KLD + j * 8],
                            Kbase + (size_t)(kv0 + row) * C3 + j * 8);
        }
        // ---- V tile transposed to [64 d][32 keys] (VGPR round-trip) ----
        for (int i = t; i < 1024; i += 128) {
            int key = i >> 5, d2 = (i & 31) * 2;
            unsigned int v = *(const unsigned int*)(Vbase + (size_t)(kv0 + key) * C3 + d2);
            Vs[d2 * VLD + key]       = (unsigned short)(v & 0xFFFFu);
            Vs[(d2 + 1) * VLD + key] = (unsigned short)(v >> 16);
        }
        wait_async0();
        __syncthreads();

        if (kv0 <= qlast) {
            // ---- S = Q K^T  (two 16x16 score tiles, K-dim = 64 = 2x32) ----
            f32x8 s[2];
#pragma unroll
            for (int nt = 0; nt < 2; ++nt) {
#pragma unroll
                for (int r = 0; r < 8; ++r) s[nt][r] = 0.0f;
                int kb = (nt * 16 + lm) * KLD;
                Frag kf0, kf1;
                kf0.q[0] = *(const uint4*)&Ks[kb + hi * 8];
                kf0.q[1] = *(const uint4*)&Ks[kb + 16 + hi * 8];
                kf1.q[0] = *(const uint4*)&Ks[kb + 32 + hi * 8];
                kf1.q[1] = *(const uint4*)&Ks[kb + 32 + 16 + hi * 8];
                s[nt] = __builtin_amdgcn_wmma_f32_16x16x32_bf16(
                    false, qf[0].v, false, kf0.v, (short)0, s[nt], false, false);
                s[nt] = __builtin_amdgcn_wmma_f32_16x16x32_bf16(
                    false, qf[1].v, false, kf1.v, (short)0, s[nt], false, false);
            }

            // ---- scale + causal mask + online softmax (per row r+8*hi) ----
#pragma unroll
            for (int r = 0; r < 8; ++r) {
                int qq = q0 + wave * 16 + r + hi * 8;
                float p0 = s[0][r] * 0.125f;
                float p1 = s[1][r] * 0.125f;
                if (kv0 + lm      > qq) p0 = -1.0e30f;
                if (kv0 + 16 + lm > qq) p1 = -1.0e30f;

                float cm = fmaxf(p0, p1);
#pragma unroll
                for (int off = 1; off < 16; off <<= 1)
                    cm = fmaxf(cm, __shfl_xor(cm, off, 32));
                float nm = fmaxf(mrow[r], cm);
                float alpha = __expf(mrow[r] - nm);
                p0 = __expf(p0 - nm);
                p1 = __expf(p1 - nm);
                float rs = p0 + p1;
#pragma unroll
                for (int off = 1; off < 16; off <<= 1)
                    rs += __shfl_xor(rs, off, 32);
                lrow[r] = lrow[r] * alpha + rs;
                mrow[r] = nm;
#pragma unroll
                for (int d = 0; d < 4; ++d) o[d][r] *= alpha;

                // stash P (bf16) for re-layout into an A fragment
                Ps[wave][(r + hi * 8) * PLD + lm]      = f2bf(p0);
                Ps[wave][(r + hi * 8) * PLD + 16 + lm] = f2bf(p1);
            }

            // ---- O += P V  (P as 16x32 A fragment, V^T rows as B fragments) ----
            Frag pf;
            pf.q[0] = *(const uint4*)&Ps[wave][lm * PLD + hi * 8];
            pf.q[1] = *(const uint4*)&Ps[wave][lm * PLD + 16 + hi * 8];
#pragma unroll
            for (int dt = 0; dt < 4; ++dt) {
                Frag vf;
                int vb = (dt * 16 + lm) * VLD;
                vf.q[0] = *(const uint4*)&Vs[vb + hi * 8];
                vf.q[1] = *(const uint4*)&Vs[vb + 16 + hi * 8];
                o[dt] = __builtin_amdgcn_wmma_f32_16x16x32_bf16(
                    false, pf.v, false, vf.v, (short)0, o[dt], false, false);
            }
        }
        __syncthreads();
    }

    // ---- normalize and store bf16 [B*T][1024] ----
#pragma unroll
    for (int dt = 0; dt < 4; ++dt)
#pragma unroll
        for (int r = 0; r < 8; ++r) {
            float v = o[dt][r] / lrow[r];
            int row = b * TSEQ + q0 + wave * 16 + r + hi * 8;
            int col = h * 64 + dt * 16 + lm;
            out[(size_t)row * CMODEL + col] = f2bf(v);
        }
}

// ---------------------------------------------------------------------------
// Host-side orchestration
// ---------------------------------------------------------------------------
extern "C" void kernel_launch(void* const* d_in, const int* in_sizes, int n_in,
                              void* d_out, int out_size, void* d_ws, size_t ws_size,
                              hipStream_t stream) {
    const float* x      = (const float*)d_in[0];
    const float* w_qkv  = (const float*)d_in[1];
    const float* w_proj = (const float*)d_in[2];
    const float* w1     = (const float*)d_in[3];
    const float* w2     = (const float*)d_in[4];
    const float* ln1_g  = (const float*)d_in[5];
    const float* ln1_b  = (const float*)d_in[6];
    const float* ln2_g  = (const float*)d_in[7];
    const float* ln2_b  = (const float*)d_in[8];
    float* outp = (float*)d_out;

    char* ws = (char*)d_ws;
    size_t off = 0;
    auto take = [&](size_t bytes) -> void* {
        void* p = ws + off;
        off += (bytes + 255) & ~(size_t)255;
        return p;
    };
    unsigned short* xn1    = (unsigned short*)take((size_t)ROWS * CMODEL * 2);
    unsigned short* wqkvT  = (unsigned short*)take((size_t)C3 * CMODEL * 2);
    unsigned short* wprojT = (unsigned short*)take((size_t)CMODEL * CMODEL * 2);
    unsigned short* w1T    = (unsigned short*)take((size_t)DFF * CMODEL * 2);
    unsigned short* w2T    = (unsigned short*)take((size_t)CMODEL * DFF * 2);
    unsigned short* qkvb   = (unsigned short*)take((size_t)ROWS * C3 * 2);
    unsigned short* attnb  = (unsigned short*)take((size_t)ROWS * CMODEL * 2);
    float*          x1     = (float*)take((size_t)ROWS * CMODEL * 4);
    unsigned short* xn2    = (unsigned short*)take((size_t)ROWS * CMODEL * 2);
    unsigned short* hbuf   = (unsigned short*)take((size_t)ROWS * DFF * 2);
    (void)ws_size; (void)in_sizes; (void)n_in; (void)out_size;

    // 1) LN1
    ln_bf16_kernel<<<ROWS, 256, 0, stream>>>(x, ln1_g, ln1_b, xn1);

    // 2) weight transpose+convert (once per launch; deterministic)
    transpose_bf16_kernel<<<dim3(C3 / 32, CMODEL / 32), 256, 0, stream>>>(w_qkv, wqkvT, CMODEL, C3);
    transpose_bf16_kernel<<<dim3(CMODEL / 32, CMODEL / 32), 256, 0, stream>>>(w_proj, wprojT, CMODEL, CMODEL);
    transpose_bf16_kernel<<<dim3(DFF / 32, CMODEL / 32), 256, 0, stream>>>(w1, w1T, CMODEL, DFF);
    transpose_bf16_kernel<<<dim3(CMODEL / 32, DFF / 32), 256, 0, stream>>>(w2, w2T, DFF, CMODEL);

    // 3) QKV = xn1 @ w_qkv   -> bf16 [4096][3072]
    gemm_bf16_wmma_kernel<<<dim3(C3 / BN, ROWS / BM), 256, 0, stream>>>(
        xn1, wqkvT, ROWS, C3, CMODEL, 0, qkvb, nullptr, nullptr);

    // 4) causal flash attention -> bf16 [4096][1024]
    attn_kernel<<<2 * 16 * (TSEQ / 64), 128, 0, stream>>>(qkvb, attnb);

    // 5) x1 = attn @ w_proj + x
    gemm_bf16_wmma_kernel<<<dim3(CMODEL / BN, ROWS / BM), 256, 0, stream>>>(
        attnb, wprojT, ROWS, CMODEL, CMODEL, 1, nullptr, x1, x);

    // 6) LN2
    ln_bf16_kernel<<<ROWS, 256, 0, stream>>>(x1, ln2_g, ln2_b, xn2);

    // 7) h = gelu(xn2 @ w1)  -> bf16 [4096][4096]
    gemm_bf16_wmma_kernel<<<dim3(DFF / BN, ROWS / BM), 256, 0, stream>>>(
        xn2, w1T, ROWS, DFF, CMODEL, 2, hbuf, nullptr, nullptr);

    // 8) out = h @ w2 + x1
    gemm_bf16_wmma_kernel<<<dim3(CMODEL / BN, ROWS / BM), 256, 0, stream>>>(
        hbuf, w2T, ROWS, CMODEL, DFF, 1, nullptr, outp, x1);
}
